// MixtralDecoderLayer_44659069944440
// MI455X (gfx1250) — compile-verified, matrix operand order checked by reading, not run on hardware
//
#include <hip/hip_runtime.h>

// ---------------------------------------------------------------------------
// Mixtral decoder layer for gfx1250 (MI455X), bf16 WMMA everywhere.
// Data movement: A tiles via per-lane async global->LDS (ASYNCcnt),
// B (weight) tiles via Tensor Data Mover descriptors (TENSORcnt).
// ---------------------------------------------------------------------------

typedef __bf16 v16bf __attribute__((ext_vector_type(16)));
typedef __bf16 v8bf  __attribute__((ext_vector_type(8)));
typedef float  v8f   __attribute__((ext_vector_type(8)));
typedef unsigned int u32x4 __attribute__((ext_vector_type(4)));
typedef int   i32x8 __attribute__((ext_vector_type(8)));
typedef int   i32x4 __attribute__((ext_vector_type(4)));

// Exact pointee type for the gfx1250 async-to-LDS builtins (verified against
// hipcc diagnostics: parameter is int __vector(4)* in AS1/AS3).
typedef int v4i_gcc __attribute__((vector_size(4 * sizeof(int))));
typedef __attribute__((address_space(1))) v4i_gcc* g_v4i_ptr;
typedef __attribute__((address_space(3))) v4i_gcc* l_v4i_ptr;

#define BM 128
#define BN 64
#define BK 32

#if defined(__has_builtin)
#if __has_builtin(__builtin_amdgcn_global_load_async_to_lds_b128)
#define HAVE_ASYNC_LDS 1
#endif
#if __has_builtin(__builtin_amdgcn_tensor_load_to_lds)
#define HAVE_TDM 1
#endif
#endif

__device__ __forceinline__ void wait_asynccnt0() {
#if defined(__has_builtin) && __has_builtin(__builtin_amdgcn_s_wait_asynccnt)
  __builtin_amdgcn_s_wait_asynccnt(0);
#else
  asm volatile("s_wait_asynccnt 0x0" ::: "memory");
#endif
}

__device__ __forceinline__ void wait_tensorcnt0() {
#if defined(__has_builtin) && __has_builtin(__builtin_amdgcn_s_wait_tensorcnt)
  __builtin_amdgcn_s_wait_tensorcnt(0);
#else
  asm volatile("s_wait_tensorcnt 0x0" ::: "memory");
#endif
}

// Build a 16-element bf16 fragment from two contiguous 8-element halves
// (16B vector loads -> ds_load_b128 / global_load_b128, no scalar packing).
__device__ __forceinline__ v16bf make_frag(const __bf16* p0, const __bf16* p1) {
  v8bf lo = *(const v8bf*)p0;
  v8bf hi = *(const v8bf*)p1;
  return __builtin_shufflevector(lo, hi, 0, 1, 2, 3, 4, 5, 6, 7,
                                 8, 9, 10, 11, 12, 13, 14, 15);
}

// Build a bf16 fragment from 16 fp32 LDS values (two 32B halves); the
// conversions lower to packed v_cvt ops.
__device__ __forceinline__ v16bf cvt_frag(const float* p0, const float* p1) {
  const float4 a0 = *(const float4*)p0;
  const float4 a1 = *(const float4*)(p0 + 4);
  const float4 b0 = *(const float4*)p1;
  const float4 b1 = *(const float4*)(p1 + 4);
  v16bf r;
  r[0]  = (__bf16)a0.x; r[1]  = (__bf16)a0.y; r[2]  = (__bf16)a0.z; r[3]  = (__bf16)a0.w;
  r[4]  = (__bf16)a1.x; r[5]  = (__bf16)a1.y; r[6]  = (__bf16)a1.z; r[7]  = (__bf16)a1.w;
  r[8]  = (__bf16)b0.x; r[9]  = (__bf16)b0.y; r[10] = (__bf16)b0.z; r[11] = (__bf16)b0.w;
  r[12] = (__bf16)b1.x; r[13] = (__bf16)b1.y; r[14] = (__bf16)b1.z; r[15] = (__bf16)b1.w;
  return r;
}

// ---------------------------------------------------------------------------
// Generic GEMM: C[M,N] = A[M,K](bf16) * B[N,K](fp32 -> bf16 in-register)ᵀ
// 256 threads = 8 waves, wave grid 4x2, each wave a 32x32 C tile.
// Double-buffered LDS. A tiles: per-lane async global->LDS (ASYNCcnt).
// B tiles: one TDM descriptor per tile (TENSORcnt), issued by thread 0.
// ---------------------------------------------------------------------------
template <bool SCALE, bool ACCUM>
__global__ __launch_bounds__(256) void gemm_bf16_kernel(
    float* __restrict__ C, const __bf16* __restrict__ A,
    const float* __restrict__ B, int M, int N, int K,
    const float* __restrict__ rowScale, int rsStride)
{
  __shared__ __bf16 sA[2][BM][BK];
  __shared__ float  sBf[2][BN][BK];

  const int tid  = threadIdx.x;
  const int lane = tid & 31;
  const int wid  = tid >> 5;
  const int wm   = wid & 3;
  const int wn   = wid >> 2;
  const int bm   = blockIdx.y * BM;
  const int bn   = blockIdx.x * BN;

  v8f acc[2][2];
#pragma unroll
  for (int i = 0; i < 2; ++i)
#pragma unroll
    for (int j = 0; j < 2; ++j)
#pragma unroll
      for (int r = 0; r < 8; ++r) acc[i][j][r] = 0.0f;

  const int rr  = lane & 15;
  const int off = (lane >> 4) * 8;   // ISA 16-bit A layout: lane-group K offset

  const int ra  = tid >> 1;            // A stage: row 0..127
  const int ca  = (tid & 1) * 16;      // A stage: col 0 or 16
  const int rb  = tid >> 2;            // B stage: row 0..63
  const int cb  = (tid & 3) * 8;       // B stage: col 0,8,16,24

  auto stageA = [&](int buf, int k0) {
    const __bf16* src = A + (size_t)(bm + ra) * K + k0 + ca;
#if HAVE_ASYNC_LDS
    __builtin_amdgcn_global_load_async_to_lds_b128(
        (g_v4i_ptr)src, (l_v4i_ptr)&sA[buf][ra][ca], 0, 0);
    __builtin_amdgcn_global_load_async_to_lds_b128(
        (g_v4i_ptr)(src + 8), (l_v4i_ptr)&sA[buf][ra][ca + 8], 0, 0);
#else
    *(uint4*)&sA[buf][ra][ca]     = *(const uint4*)src;
    *(uint4*)&sA[buf][ra][ca + 8] = *(const uint4*)(src + 8);
#endif
  };

  auto stageB = [&](int buf, int k0) {
#if HAVE_TDM
    if (tid == 0) {
      // 2D Tensor DMA descriptor: tile_dim0=BK (contiguous fp32), tile_dim1=BN
      // rows, tensor_dim0_stride = K, source B + bn*K + k0.
      const size_t gaddr = (size_t)(B + (size_t)bn * K + k0);
      u32x4 g0;
      g0.x = 1u;                                          // count=1, load
      g0.y = (unsigned)(size_t)&sBf[buf][0][0];           // lds_addr (bytes)
      g0.z = (unsigned)(gaddr & 0xFFFFFFFFu);             // global_addr lo
      g0.w = (unsigned)((gaddr >> 32) & 0x1FFFFFFu) | (2u << 30);  // hi | type=2
      i32x8 g1;
      g1[0] = 0x20000;                                    // data_size = 4B
      g1[1] = (int)(((unsigned)K & 0xFFFFu) << 16);       // tensor_dim0 lo16
      g1[2] = (int)((((unsigned)K >> 16) & 0xFFFFu) |
                    (((unsigned)N & 0xFFFFu) << 16));     // dim0 hi | dim1 lo
      g1[3] = (int)((((unsigned)N >> 16) & 0xFFFFu) |
                    ((unsigned)BK << 16));                // dim1 hi | tile_dim0
      g1[4] = BN;                                         // tile_dim1 (tile_dim2=0)
      g1[5] = K;                                          // tensor_dim0_stride lo32
      g1[6] = 0;
      g1[7] = 0;
      const i32x4 gz = {0, 0, 0, 0};
#if __clang_major__ >= 23
      const i32x8 gz8 = {0, 0, 0, 0, 0, 0, 0, 0};
      __builtin_amdgcn_tensor_load_to_lds(g0, g1, gz, gz, gz8, 0);
#else
      __builtin_amdgcn_tensor_load_to_lds(g0, g1, gz, gz, 0);
#endif
    }
#else
    *(float4*)&sBf[buf][rb][cb]     = *(const float4*)(B + (size_t)(bn + rb) * K + k0 + cb);
    *(float4*)&sBf[buf][rb][cb + 4] = *(const float4*)(B + (size_t)(bn + rb) * K + k0 + cb + 4);
#endif
  };

  auto waitStage = [&]() {
#if HAVE_ASYNC_LDS
    wait_asynccnt0();
#endif
#if HAVE_TDM
    if (tid == 0) wait_tensorcnt0();
#endif
  };

  // Prologue: stage tile 0.
  stageA(0, 0);
  stageB(0, 0);
  waitStage();
  __syncthreads();

  const int nk = K / BK;
  for (int kt = 0; kt < nk; ++kt) {
    const int cur = kt & 1;
    const int nxt = cur ^ 1;
    // Stage the next tile into the alternate buffer (overlaps with compute).
    if (kt + 1 < nk) {
      stageA(nxt, (kt + 1) * BK);
      stageB(nxt, (kt + 1) * BK);
    }

    v16bf af[2], bfm[2];
#pragma unroll
    for (int t = 0; t < 2; ++t) {
      const int row = wm * 32 + t * 16 + rr;
      af[t]  = make_frag(&sA[cur][row][off], &sA[cur][row][16 + off]);
      const int col = wn * 32 + t * 16 + rr;
      bfm[t] = cvt_frag(&sBf[cur][col][off], &sBf[cur][col][16 + off]);
    }
#pragma unroll
    for (int i = 0; i < 2; ++i)
#pragma unroll
      for (int j = 0; j < 2; ++j)
        acc[i][j] = __builtin_amdgcn_wmma_f32_16x16x32_bf16(
            false, af[i], false, bfm[j], (short)0, acc[i][j], false, false);

    if (kt + 1 < nk) waitStage();
    __syncthreads();
  }

  // Epilogue. C layout: VGPR r holds M=r (lanes 0-15) / M=r+8 (lanes 16-31).
  const int g8 = (lane >> 4) * 8;
  const int cn = lane & 15;
#pragma unroll
  for (int i = 0; i < 2; ++i)
#pragma unroll
    for (int j = 0; j < 2; ++j)
#pragma unroll
      for (int r = 0; r < 8; ++r) {
        const int row = bm + wm * 32 + i * 16 + r + g8;
        const int col = bn + wn * 32 + j * 16 + cn;
        float v = acc[i][j][r];
        if (SCALE) v *= rowScale[(size_t)row * rsStride];
        const size_t idx = (size_t)row * N + col;
        if (ACCUM) v += C[idx];
        C[idx] = v;
      }
}

// ---------------------------------------------------------------------------
// RMSNorm (optionally fused residual add). One block per row, D = 2048.
// ---------------------------------------------------------------------------
__global__ __launch_bounds__(256) void rmsnorm_kernel(
    const float* __restrict__ x, const float* __restrict__ addend,
    const float* __restrict__ w, __bf16* __restrict__ outb,
    float* __restrict__ outResid)
{
  const int D = 2048;
  __shared__ float red[8];
  __shared__ float stot;
  const int row = blockIdx.x;
  const float* xr = x + (size_t)row * D;
  float vals[8];
  float ss = 0.f;
#pragma unroll
  for (int i = 0; i < 8; ++i) {
    const int c = i * 256 + threadIdx.x;
    float v = xr[c];
    if (addend) v += addend[(size_t)row * D + c];
    vals[i] = v;
    ss += v * v;
  }
#pragma unroll
  for (int m = 16; m >= 1; m >>= 1) ss += __shfl_xor(ss, m, 32);
  if ((threadIdx.x & 31) == 0) red[threadIdx.x >> 5] = ss;
  __syncthreads();
  if (threadIdx.x == 0) {
    float t = 0.f;
#pragma unroll
    for (int i = 0; i < 8; ++i) t += red[i];
    stot = rsqrtf(t / (float)D + 1e-5f);
  }
  __syncthreads();
  const float rs = stot;
#pragma unroll
  for (int i = 0; i < 8; ++i) {
    const int c = i * 256 + threadIdx.x;
    const float v = vals[i];
    if (outResid) outResid[(size_t)row * D + c] = v;
    outb[(size_t)row * D + c] = (__bf16)(v * rs * w[c]);
  }
}

// ---------------------------------------------------------------------------
// RoPE (HF rotate_half) fp32 -> bf16. grid (S, NH), block 64 (= HD/2).
// ---------------------------------------------------------------------------
__global__ __launch_bounds__(64) void rope_kernel(const float* __restrict__ x,
                                                  __bf16* __restrict__ y)
{
  const int t = blockIdx.x;
  const int h = blockIdx.y;
  const int i = threadIdx.x;
  const float inv = __powf(10000.f, -2.0f * (float)i / 128.0f);
  const float ang = (float)t * inv;
  const float c = __cosf(ang), s = __sinf(ang);
  const size_t base = (size_t)t * 2048 + h * 128;
  const float x1 = x[base + i];
  const float x2 = x[base + 64 + i];
  y[base + i]      = (__bf16)(x1 * c - x2 * s);
  y[base + 64 + i] = (__bf16)(x2 * c + x1 * s);
}

// V transpose: vT[(h*128+d)*S + t] = v[t*2048 + h*128 + d], fp32 -> bf16.
__global__ void transpose_v_kernel(__bf16* __restrict__ vT,
                                   const float* __restrict__ v)
{
  const size_t i = (size_t)blockIdx.x * 256 + threadIdx.x;  // i = d*S + t
  const int d = (int)(i >> 11);     // / 2048
  const int t = (int)(i & 2047);
  vT[i] = (__bf16)v[(size_t)t * 2048 + d];
}

// ---------------------------------------------------------------------------
// Flash attention, causal. One wave per (head, 16-query tile).
// ---------------------------------------------------------------------------
__global__ __launch_bounds__(32) void attn_kernel(
    __bf16* __restrict__ outb, const __bf16* __restrict__ qb,
    const __bf16* __restrict__ kb, const __bf16* __restrict__ vT)
{
  __shared__ __bf16 sP[16][32];
  const int lane = threadIdx.x;
  const int h  = blockIdx.x & 15;          // NH = 16
  const int q0 = (blockIdx.x >> 4) * 16;
  const int rr  = lane & 15;
  const int off = (lane >> 4) * 8;
  const int g8  = (lane >> 4) * 8;
  const float scale = 0.08838834764831845f;  // 1/sqrt(128)

  v16bf qf[4];
  {
    const __bf16* qrow = qb + (size_t)(q0 + rr) * 2048 + h * 128;
#pragma unroll
    for (int c = 0; c < 4; ++c)
      qf[c] = make_frag(qrow + c * 32 + off, qrow + c * 32 + 16 + off);
  }

  v8f o[8];
#pragma unroll
  for (int d = 0; d < 8; ++d)
#pragma unroll
    for (int r = 0; r < 8; ++r) o[d][r] = 0.0f;
  float mrow[8], lrow[8];
#pragma unroll
  for (int r = 0; r < 8; ++r) { mrow[r] = -1e30f; lrow[r] = 0.0f; }

  const int nblk = (q0 + 16 + 31) / 32;
  for (int jb = 0; jb < nblk; ++jb) {
    const int kbase = jb * 32;

    v8f s[2];
#pragma unroll
    for (int t = 0; t < 2; ++t)
#pragma unroll
      for (int r = 0; r < 8; ++r) s[t][r] = 0.0f;
#pragma unroll
    for (int t = 0; t < 2; ++t) {
      const __bf16* krow = kb + (size_t)(kbase + t * 16 + rr) * 2048 + h * 128;
#pragma unroll
      for (int c = 0; c < 4; ++c) {
        v16bf kf = make_frag(krow + c * 32 + off, krow + c * 32 + 16 + off);
        s[t] = __builtin_amdgcn_wmma_f32_16x16x32_bf16(
            false, qf[c], false, kf, (short)0, s[t], false, false);
      }
    }

#pragma unroll
    for (int r = 0; r < 8; ++r) {
      const int qi = q0 + r + g8;
      float a = s[0][r] * scale;
      if (kbase + rr > qi) a = -1e30f;
      float b = s[1][r] * scale;
      if (kbase + 16 + rr > qi) b = -1e30f;
      float mx = fmaxf(a, b);
#pragma unroll
      for (int msk = 8; msk >= 1; msk >>= 1) mx = fmaxf(mx, __shfl_xor(mx, msk, 32));
      const float mnew = fmaxf(mrow[r], mx);
      const float p0 = __expf(a - mnew);
      const float p1 = __expf(b - mnew);
      float rowsum = p0 + p1;
#pragma unroll
      for (int msk = 8; msk >= 1; msk >>= 1) rowsum += __shfl_xor(rowsum, msk, 32);
      const float f = __expf(mrow[r] - mnew);
      lrow[r] = lrow[r] * f + rowsum;
      mrow[r] = mnew;
#pragma unroll
      for (int d = 0; d < 8; ++d) o[d][r] *= f;
      sP[r + g8][rr]      = (__bf16)p0;
      sP[r + g8][16 + rr] = (__bf16)p1;
    }
    __syncthreads();

    const v16bf pf = make_frag(&sP[rr][off], &sP[rr][16 + off]);
#pragma unroll
    for (int d = 0; d < 8; ++d) {
      // vT row = head dim (h*128 + d*16 + rr), columns = keys (contiguous).
      const __bf16* vrow = vT + (size_t)(h * 128 + d * 16 + rr) * 2048 + kbase;
      v16bf vf = make_frag(vrow + off, vrow + 16 + off);
      o[d] = __builtin_amdgcn_wmma_f32_16x16x32_bf16(
          false, pf, false, vf, (short)0, o[d], false, false);
    }
    __syncthreads();
  }

#pragma unroll
  for (int d = 0; d < 8; ++d)
#pragma unroll
    for (int r = 0; r < 8; ++r) {
      const size_t idx = (size_t)(q0 + r + g8) * 2048 + h * 128 + d * 16 + rr;
      outb[idx] = (__bf16)(o[d][r] / lrow[r]);
    }
}

// ---------------------------------------------------------------------------
// MoE gate: softmax over 8 experts, top-2, renormalized combine weights.
// ---------------------------------------------------------------------------
__global__ __launch_bounds__(64) void gate_kernel(const __bf16* __restrict__ xn,
                                                  const float* __restrict__ gw,
                                                  float* __restrict__ cw)
{
  const int t = blockIdx.x * 64 + threadIdx.x;
  float logits[8];
  const __bf16* xr = xn + (size_t)t * 2048;
#pragma unroll
  for (int e = 0; e < 8; ++e) {
    float acc = 0.f;
    const float* g = gw + (size_t)e * 2048;
    for (int d = 0; d < 2048; ++d) acc += (float)xr[d] * g[d];
    logits[e] = acc;
  }
  float mx = logits[0];
#pragma unroll
  for (int e = 1; e < 8; ++e) mx = fmaxf(mx, logits[e]);
  float p[8];
#pragma unroll
  for (int e = 0; e < 8; ++e) p[e] = __expf(logits[e] - mx);
  int i1 = 0;
#pragma unroll
  for (int e = 1; e < 8; ++e) if (p[e] > p[i1]) i1 = e;
  int i2 = (i1 == 0) ? 1 : 0;
#pragma unroll
  for (int e = 0; e < 8; ++e) if (e != i1 && p[e] > p[i2]) i2 = e;
  const float s = p[i1] + p[i2];
#pragma unroll
  for (int e = 0; e < 8; ++e) cw[(size_t)t * 8 + e] = 0.f;
  cw[(size_t)t * 8 + i1] = p[i1] / s;
  cw[(size_t)t * 8 + i2] = p[i2] / s;
}

__global__ void silumul_kernel(__bf16* __restrict__ act,
                               const float* __restrict__ h1,
                               const float* __restrict__ h3, size_t n)
{
  const size_t i = (size_t)blockIdx.x * blockDim.x + threadIdx.x;
  if (i < n) {
    const float a = h1[i];
    const float sg = a / (1.f + __expf(-a));
    act[i] = (__bf16)(sg * h3[i]);
  }
}

// ---------------------------------------------------------------------------
// Host orchestration.
// ---------------------------------------------------------------------------
extern "C" void kernel_launch(void* const* d_in, const int* in_sizes, int n_in,
                              void* d_out, int out_size, void* d_ws, size_t ws_size,
                              hipStream_t stream) {
  const int S = 2048, D = 2048, NH = 16, E = 8, I = 4096;
  (void)in_sizes; (void)n_in; (void)out_size; (void)ws_size;

  const float* hidden = (const float*)d_in[0];
  const float* wq     = (const float*)d_in[1];
  const float* wk     = (const float*)d_in[2];
  const float* wv     = (const float*)d_in[3];
  const float* wo     = (const float*)d_in[4];
  const float* ln1    = (const float*)d_in[5];
  const float* ln2    = (const float*)d_in[6];
  const float* gate_w = (const float*)d_in[7];
  const float* w1     = (const float*)d_in[8];
  const float* w2     = (const float*)d_in[9];
  const float* w3     = (const float*)d_in[10];

  float* out_hs    = (float*)d_out;
  float* out_resid = out_hs + (size_t)S * D;

  char* ws = (char*)d_ws;
  auto alloc = [&](size_t bytes) -> char* {
    char* p = ws;
    ws += (bytes + 255) & ~(size_t)255;
    return p;
  };
  __bf16* xn1   = (__bf16*)alloc((size_t)S * D * 2);
  float*  qf    = (float*) alloc((size_t)S * D * 4);
  float*  kf    = (float*) alloc((size_t)S * D * 4);
  float*  vf    = (float*) alloc((size_t)S * D * 4);
  __bf16* qb    = (__bf16*)alloc((size_t)S * D * 2);
  __bf16* kb    = (__bf16*)alloc((size_t)S * D * 2);
  __bf16* vT    = (__bf16*)alloc((size_t)S * D * 2);
  __bf16* attnb = (__bf16*)alloc((size_t)S * D * 2);
  __bf16* xn2   = (__bf16*)alloc((size_t)S * D * 2);
  float*  cw    = (float*) alloc((size_t)S * E * 4);
  float*  h1    = (float*) alloc((size_t)S * I * 4);
  float*  h3    = (float*) alloc((size_t)S * I * 4);
  __bf16* act   = (__bf16*)alloc((size_t)S * I * 2);
  float*  attnproj = qf;  // reuse q fp32 buffer after RoPE consumed it

  // 1) RMSNorm1 -> bf16
  rmsnorm_kernel<<<S, 256, 0, stream>>>(hidden, nullptr, ln1, xn1, nullptr);

  // 2) QKV projections
  const dim3 gProj(D / BN, S / BM);
  gemm_bf16_kernel<false, false><<<gProj, 256, 0, stream>>>(qf, xn1, wq, S, D, D, nullptr, 0);
  gemm_bf16_kernel<false, false><<<gProj, 256, 0, stream>>>(kf, xn1, wk, S, D, D, nullptr, 0);
  gemm_bf16_kernel<false, false><<<gProj, 256, 0, stream>>>(vf, xn1, wv, S, D, D, nullptr, 0);

  // 3) RoPE on Q/K, transpose V to [dim][seq] bf16
  rope_kernel<<<dim3(S, NH), 64, 0, stream>>>(qf, qb);
  rope_kernel<<<dim3(S, NH), 64, 0, stream>>>(kf, kb);
  transpose_v_kernel<<<(int)(((size_t)S * D) / 256), 256, 0, stream>>>(vT, vf);

  // 4) Causal flash attention
  attn_kernel<<<NH * (S / 16), 32, 0, stream>>>(attnb, qb, kb, vT);

  // 5) Output projection
  gemm_bf16_kernel<false, false><<<gProj, 256, 0, stream>>>(attnproj, attnb, wo, S, D, D, nullptr, 0);

  // 6) Fused residual add + RMSNorm2 (writes residual output)
  rmsnorm_kernel<<<S, 256, 0, stream>>>(hidden, attnproj, ln2, xn2, out_resid);

  // 7) Routing
  gate_kernel<<<S / 64, 64, 0, stream>>>(xn2, gate_w, cw);

  // 8) MoE expert FFNs, weighted-accumulated into out_hs
  (void)hipMemsetAsync(out_hs, 0, (size_t)S * D * 4, stream);
  const dim3 gUp(I / BN, S / BM);
  for (int e = 0; e < E; ++e) {
    const float* w1e = w1 + (size_t)e * I * D;
    const float* w3e = w3 + (size_t)e * I * D;
    const float* w2e = w2 + (size_t)e * D * I;
    gemm_bf16_kernel<false, false><<<gUp, 256, 0, stream>>>(h1, xn2, w1e, S, I, D, nullptr, 0);
    gemm_bf16_kernel<false, false><<<gUp, 256, 0, stream>>>(h3, xn2, w3e, S, I, D, nullptr, 0);
    silumul_kernel<<<(int)(((size_t)S * I + 255) / 256), 256, 0, stream>>>(act, h1, h3, (size_t)S * I);
    gemm_bf16_kernel<true, true><<<gProj, 256, 0, stream>>>(out_hs, act, w2e, S, D, I, cw + e, E);
  }
}
